// RNNModel_27891517620655
// MI455X (gfx1250) — compile-verified, matrix operand order checked by reading, not run on hardware
//
#include <hip/hip_runtime.h>

typedef __attribute__((ext_vector_type(16))) __bf16 v16bf;
typedef __attribute__((ext_vector_type(8)))  __bf16 v8bf;
typedef __attribute__((ext_vector_type(8)))  float  v8f;

namespace {
constexpr int S = 40, N = 256, E = 1024, H = 1024, L = 2;
constexpr int H3 = 3 * H;          // 3072
constexpr int AK = E + L * H;      // 3072 (attention input width)
constexpr int CK = E + H;          // 2048 (combine input width)
}

// ---------------- elementwise kernels ----------------

__global__ void k_cvt_bf16(const float* __restrict__ src, __bf16* __restrict__ dst,
                           size_t n) {
  size_t i = (size_t)blockIdx.x * blockDim.x + threadIdx.x;
  if (i < n) dst[i] = (__bf16)src[i];
}

__global__ void k_copy_f32(const float* __restrict__ src, float* __restrict__ dst,
                           size_t n) {
  size_t i = (size_t)blockIdx.x * blockDim.x + threadIdx.x;
  if (i < n) dst[i] = src[i];
}

__global__ void k_embed(const int* __restrict__ tokens, const float* __restrict__ emb,
                        __bf16* __restrict__ xbf) {
  size_t i = (size_t)blockIdx.x * blockDim.x + threadIdx.x;
  if (i >= (size_t)S * N * E) return;
  int row = (int)(i / E);            // s*N + n
  int e   = (int)(i % E);
  int tok = tokens[row];
  xbf[i] = (__bf16)emb[(size_t)tok * E + e];
}

__global__ void k_init_h(const float* __restrict__ src, float* __restrict__ h,
                         __bf16* __restrict__ hbf, int n) {
  int i = blockIdx.x * blockDim.x + threadIdx.x;
  if (i >= n) return;
  float v = src[i];
  h[i] = v;
  hbf[i] = (__bf16)v;
}

// PyTorch GRU gates, order [r, z, n]. Updates h (f32) in place, mirrors to bf16,
// optionally also writes a sequence slot (bf16 and/or f32).
__global__ void k_gru_gates(const float* __restrict__ gi, const float* __restrict__ gh,
                            const float* __restrict__ bih, const float* __restrict__ bhh,
                            float* __restrict__ h, __bf16* __restrict__ hbf,
                            __bf16* __restrict__ out_bf, float* __restrict__ out_f32) {
  int i = blockIdx.x * blockDim.x + threadIdx.x;
  if (i >= N * H) return;
  int n = i / H, j = i % H;
  size_t b = (size_t)n * H3 + j;
  float ir  = gi[b]           + bih[j];
  float iz  = gi[b + H]       + bih[H + j];
  float in_ = gi[b + 2 * H]   + bih[2 * H + j];
  float hr  = gh[b]           + bhh[j];
  float hz  = gh[b + H]       + bhh[H + j];
  float hn  = gh[b + 2 * H]   + bhh[2 * H + j];
  float r  = 1.f / (1.f + expf(-(ir + hr)));
  float z  = 1.f / (1.f + expf(-(iz + hz)));
  float nn = tanhf(in_ + r * hn);
  float hv = (1.f - z) * nn + z * h[i];
  h[i]   = hv;
  hbf[i] = (__bf16)hv;
  if (out_bf)  out_bf[i]  = (__bf16)hv;
  if (out_f32) out_f32[i] = hv;
}

__global__ void k_softmax(const float* __restrict__ logits, const float* __restrict__ bias,
                          float* __restrict__ aw) {
  int n = blockIdx.x * blockDim.x + threadIdx.x;
  if (n >= N) return;
  float v[S];
  float mx = -1e30f;
#pragma unroll
  for (int s = 0; s < S; ++s) {
    float t = logits[n * S + s] + bias[s];
    v[s] = t;
    mx = fmaxf(mx, t);
  }
  float sum = 0.f;
#pragma unroll
  for (int s = 0; s < S; ++s) { v[s] = expf(v[s] - mx); sum += v[s]; }
  float inv = 1.f / sum;
#pragma unroll
  for (int s = 0; s < S; ++s) aw[n * S + s] = v[s] * inv;
}

// applied[n][j] = sum_s aw[n][s] * encout[s][n][j]
__global__ void k_attn_apply(const float* __restrict__ aw, const float* __restrict__ encout,
                             float* __restrict__ applied) {
  int i = blockIdx.x * blockDim.x + threadIdx.x;
  if (i >= N * H) return;
  int n = i / H;
  float acc = 0.f;
#pragma unroll 8
  for (int s = 0; s < S; ++s)
    acc = fmaf(aw[n * S + s], encout[(size_t)s * N * H + i], acc);
  applied[i] = acc;
}

// attn_in[n] = [ emb_t[n] (E) | h0[n] (H) | h1[n] (H) ]   bf16
__global__ void k_build_attn_in(const __bf16* __restrict__ xt, const __bf16* __restrict__ hbf,
                                __bf16* __restrict__ out) {
  int i = blockIdx.x * blockDim.x + threadIdx.x;
  if (i >= N * AK) return;
  int n = i / AK, c = i % AK;
  __bf16 v;
  if (c < E) {
    v = xt[(size_t)n * E + c];
  } else {
    int l = (c - E) / H, j = (c - E) % H;
    v = hbf[(size_t)l * N * H + (size_t)n * H + j];
  }
  out[i] = v;
}

// comb_in[n] = [ emb_t[n] (E) | applied[n] (H) ]   bf16
__global__ void k_build_comb_in(const __bf16* __restrict__ xt, const float* __restrict__ applied,
                                __bf16* __restrict__ out) {
  int i = blockIdx.x * blockDim.x + threadIdx.x;
  if (i >= N * CK) return;
  int n = i / CK, c = i % CK;
  out[i] = (c < E) ? xt[(size_t)n * E + c]
                   : (__bf16)applied[(size_t)n * H + (c - E)];
}

// xcur_bf = bf16(relu(comb_out + comb_b))
__global__ void k_relu_cvt(const float* __restrict__ comb_out, const float* __restrict__ comb_b,
                           __bf16* __restrict__ xcur_bf) {
  int i = blockIdx.x * blockDim.x + threadIdx.x;
  if (i >= N * H) return;
  int j = i % H;
  xcur_bf[i] = (__bf16)fmaxf(comb_out[i] + comb_b[j], 0.f);
}

// ---------------- WMMA GEMM: C(M,Nc) = A(M,K) * W(Nc,K)^T ----------------
// A, W bf16 row-major; C f32 row-major.
// Block: 256 threads = 8 waves; waves tiled 2(M) x 4(N); each wave -> 16x64 strip.

__device__ inline v16bf frag_from(const __bf16* __restrict__ p) {
  v8bf lo = *(const v8bf*)p;
  v8bf hi = *(const v8bf*)(p + 16);
  return __builtin_shufflevector(lo, hi,
      0, 1, 2, 3, 4, 5, 6, 7, 8, 9, 10, 11, 12, 13, 14, 15);
}

__device__ inline v16bf zero_frag() {
  v16bf z;
#pragma unroll
  for (int i = 0; i < 16; ++i) z[i] = (__bf16)0.f;
  return z;
}

// Fast path: M%32==0, Nc%256==0, K%32==0. Branch-free inner loop.
__global__ __launch_bounds__(256) void k_gemm_bf16_fast(
    const __bf16* __restrict__ A, const __bf16* __restrict__ W,
    float* __restrict__ C, int M, int Nc, int K) {
  const int lane = threadIdx.x & 31;
  const int wave = threadIdx.x >> 5;
  const int m0 = blockIdx.y * 32 + (wave & 1) * 16;
  const int n0 = blockIdx.x * 256 + (wave >> 1) * 64;
  const int l = lane & 15, sel = lane >> 4;

  v8f acc[4];
#pragma unroll
  for (int t = 0; t < 4; ++t)
#pragma unroll
    for (int i = 0; i < 8; ++i) acc[t][i] = 0.f;

  // A fragment: lane l (both halves) reads row m0+l; sel picks K sub-block.
  const __bf16* arow = A + (size_t)(m0 + l) * K + sel * 8;
  // B fragments: lane l reads weight row n0 + t*16 + l.
  const __bf16* wrow0 = W + (size_t)(n0 + l) * K + sel * 8;
  const size_t wstep = (size_t)16 * K;   // next 16-column tile

  for (int k0 = 0; k0 < K; k0 += 32) {
    v16bf af = frag_from(arow + k0);
    v16bf b0 = frag_from(wrow0 + k0);
    v16bf b1 = frag_from(wrow0 + wstep + k0);
    v16bf b2 = frag_from(wrow0 + 2 * wstep + k0);
    v16bf b3 = frag_from(wrow0 + 3 * wstep + k0);
    acc[0] = __builtin_amdgcn_wmma_f32_16x16x32_bf16(
        false, af, false, b0, (short)0, acc[0], false, false);
    acc[1] = __builtin_amdgcn_wmma_f32_16x16x32_bf16(
        false, af, false, b1, (short)0, acc[1], false, false);
    acc[2] = __builtin_amdgcn_wmma_f32_16x16x32_bf16(
        false, af, false, b2, (short)0, acc[2], false, false);
    acc[3] = __builtin_amdgcn_wmma_f32_16x16x32_bf16(
        false, af, false, b3, (short)0, acc[3], false, false);
  }

  // C layout: lane l<16 -> rows m0+0..7, lane l+16 -> rows m0+8..15; col n0+..+l
#pragma unroll
  for (int t = 0; t < 4; ++t) {
    int n = n0 + t * 16 + l;
#pragma unroll
    for (int i = 0; i < 8; ++i)
      C[(size_t)(m0 + sel * 8 + i) * Nc + n] = acc[t][i];
  }
}

// Ragged path (per-lane Nc edge), used only for the tiny attention-logits GEMM.
__global__ __launch_bounds__(256) void k_gemm_bf16_ragged(
    const __bf16* __restrict__ A, const __bf16* __restrict__ W,
    float* __restrict__ C, int M, int Nc, int K) {
  const int lane = threadIdx.x & 31;
  const int wave = threadIdx.x >> 5;
  const int m0 = blockIdx.y * 32 + (wave & 1) * 16;
  const int n0 = blockIdx.x * 256 + (wave >> 1) * 64;
  if (m0 >= M || n0 >= Nc) return;               // wave-uniform exit
  const int l = lane & 15, sel = lane >> 4;

  v8f acc[4];
#pragma unroll
  for (int t = 0; t < 4; ++t)
#pragma unroll
    for (int i = 0; i < 8; ++i) acc[t][i] = 0.f;

  const __bf16* arow = A + (size_t)(m0 + l) * K + sel * 8;

  for (int k0 = 0; k0 < K; k0 += 32) {
    v16bf af = frag_from(arow + k0);
#pragma unroll
    for (int t = 0; t < 4; ++t) {
      if (n0 + t * 16 < Nc) {                    // wave-uniform
        int n = n0 + t * 16 + l;
        v16bf bfrag = (n < Nc)
            ? frag_from(W + (size_t)n * K + k0 + sel * 8)
            : zero_frag();
        acc[t] = __builtin_amdgcn_wmma_f32_16x16x32_bf16(
            false, af, false, bfrag, (short)0, acc[t], false, false);
      }
    }
  }

#pragma unroll
  for (int t = 0; t < 4; ++t) {
    int n = n0 + t * 16 + l;
    if (n >= Nc) continue;
#pragma unroll
    for (int i = 0; i < 8; ++i)
      C[(size_t)(m0 + sel * 8 + i) * Nc + n] = acc[t][i];
  }
}

// ---------------- host orchestration ----------------

static inline int cdiv(long long a, long long b) { return (int)((a + b - 1) / b); }

extern "C" void kernel_launch(void* const* d_in, const int* in_sizes, int n_in,
                              void* d_out, int out_size, void* d_ws, size_t ws_size,
                              hipStream_t stream) {
  (void)in_sizes; (void)n_in; (void)out_size; (void)ws_size;

  const int*   tokens   = (const int*)  d_in[0];
  const float* hidden   = (const float*)d_in[1];
  const float* emb      = (const float*)d_in[2];
  const float* enc_wih  = (const float*)d_in[3];
  const float* enc_whh  = (const float*)d_in[4];
  const float* enc_bih  = (const float*)d_in[5];
  const float* enc_bhh  = (const float*)d_in[6];
  const float* dec_wih  = (const float*)d_in[7];
  const float* dec_whh  = (const float*)d_in[8];
  const float* dec_bih  = (const float*)d_in[9];
  const float* dec_bhh  = (const float*)d_in[10];
  const float* attn_w   = (const float*)d_in[11];
  const float* attn_b   = (const float*)d_in[12];
  const float* comb_w   = (const float*)d_in[13];
  const float* comb_b   = (const float*)d_in[14];
  float* out = (float*)d_out;

  // ---- workspace carve (≈280 MB) ----
  char* p = (char*)d_ws;
  auto alloc = [&](size_t bytes) -> void* {
    void* r = (void*)p;
    p += (bytes + 255) & ~(size_t)255;
    return r;
  };
  __bf16* xbf     = (__bf16*)alloc((size_t)S * N * E * 2);     // embedded tokens
  __bf16* seqbf   = (__bf16*)alloc((size_t)S * N * H * 2);     // encoder layer output (bf16)
  float*  encout  = (float*) alloc((size_t)S * N * H * 4);     // final encoder outputs (f32)
  float*  gi_all  = (float*) alloc((size_t)S * N * H3 * 4);    // batched input-gate GEMM
  __bf16* wencih  = (__bf16*)alloc((size_t)L * H3 * E * 2);
  __bf16* wenchh  = (__bf16*)alloc((size_t)L * H3 * H * 2);
  __bf16* wdecih  = (__bf16*)alloc((size_t)L * H3 * H * 2);
  __bf16* wdechh  = (__bf16*)alloc((size_t)L * H3 * H * 2);
  __bf16* wattn   = (__bf16*)alloc((size_t)S * AK * 2);
  __bf16* wcomb   = (__bf16*)alloc((size_t)H * CK * 2);
  float*  hbuf    = (float*) alloc((size_t)L * N * H * 4);     // hidden carries (f32)
  __bf16* hbf     = (__bf16*)alloc((size_t)L * N * H * 2);     // hidden carries (bf16)
  float*  gh      = (float*) alloc((size_t)N * H3 * 4);
  float*  gi      = (float*) alloc((size_t)N * H3 * 4);
  __bf16* attn_in = (__bf16*)alloc((size_t)N * AK * 2);
  float*  logits  = (float*) alloc((size_t)N * S * 4);
  float*  aw      = (float*) alloc((size_t)N * S * 4);
  float*  applied = (float*) alloc((size_t)N * H * 4);
  __bf16* comb_in = (__bf16*)alloc((size_t)N * CK * 2);
  float*  comb_o  = (float*) alloc((size_t)N * H * 4);
  __bf16* xcur_bf = (__bf16*)alloc((size_t)N * H * 2);

#define ELEM(kern, n, ...) \
  kern<<<cdiv((long long)(n), 256), 256, 0, stream>>>(__VA_ARGS__)
#define GEMMF(A_, W_, C_, M_, Nc_, K_) \
  k_gemm_bf16_fast<<<dim3(cdiv((Nc_), 256), cdiv((M_), 32)), 256, 0, stream>>>((A_), (W_), (C_), (M_), (Nc_), (K_))
#define GEMMR(A_, W_, C_, M_, Nc_, K_) \
  k_gemm_bf16_ragged<<<dim3(cdiv((Nc_), 256), cdiv((M_), 32)), 256, 0, stream>>>((A_), (W_), (C_), (M_), (Nc_), (K_))

  // ---- one-time (per launch) conversions ----
  ELEM(k_cvt_bf16, (size_t)L * H3 * E, enc_wih, wencih, (size_t)L * H3 * E);
  ELEM(k_cvt_bf16, (size_t)L * H3 * H, enc_whh, wenchh, (size_t)L * H3 * H);
  ELEM(k_cvt_bf16, (size_t)L * H3 * H, dec_wih, wdecih, (size_t)L * H3 * H);
  ELEM(k_cvt_bf16, (size_t)L * H3 * H, dec_whh, wdechh, (size_t)L * H3 * H);
  ELEM(k_cvt_bf16, (size_t)S * AK,     attn_w,  wattn,  (size_t)S * AK);
  ELEM(k_cvt_bf16, (size_t)H * CK,     comb_w,  wcomb,  (size_t)H * CK);
  ELEM(k_embed,    (size_t)S * N * E,  tokens,  emb, xbf);

  // ---- encoder: 2 layers ----
  for (int l = 0; l < L; ++l) {
    const __bf16* in_bf = (l == 0) ? xbf : seqbf;
    // batched input-gate GEMM over all timesteps: (S*N, 3H)
    GEMMF(in_bf, wencih + (size_t)l * H3 * E, gi_all, S * N, H3, E);
    ELEM(k_init_h, N * H, hidden + (size_t)l * N * H,
         hbuf + (size_t)l * N * H, hbf + (size_t)l * N * H, N * H);
    for (int s = 0; s < S; ++s) {
      GEMMF(hbf + (size_t)l * N * H, wenchh + (size_t)l * H3 * H, gh, N, H3, H);
      __bf16* obf = (l == 0) ? (seqbf + (size_t)s * N * H) : (__bf16*)nullptr;
      float*  of  = (l == L - 1) ? (encout + (size_t)s * N * H) : (float*)nullptr;
      ELEM(k_gru_gates, N * H, gi_all + (size_t)s * N * H3, gh,
           enc_bih + (size_t)l * H3, enc_bhh + (size_t)l * H3,
           hbuf + (size_t)l * N * H, hbf + (size_t)l * N * H, obf, of);
    }
  }

  // ---- attention decoder ----
  for (int t = 0; t < S; ++t) {
    const __bf16* xt = xbf + (size_t)t * N * E;
    ELEM(k_build_attn_in, N * AK, xt, hbf, attn_in);
    GEMMR(attn_in, wattn, logits, N, S, AK);                // ragged Nc = 40
    ELEM(k_softmax, N, logits, attn_b, aw);
    ELEM(k_attn_apply, N * H, aw, encout, applied);
    ELEM(k_build_comb_in, N * CK, xt, applied, comb_in);
    GEMMF(comb_in, wcomb, comb_o, N, H, CK);
    ELEM(k_relu_cvt, N * H, comb_o, comb_b, xcur_bf);
    // decoder layer 0: input xcur
    GEMMF(xcur_bf, wdecih, gi, N, H3, H);
    GEMMF(hbf, wdechh, gh, N, H3, H);
    ELEM(k_gru_gates, N * H, gi, gh, dec_bih, dec_bhh,
         hbuf, hbf, (__bf16*)nullptr, (float*)nullptr);
    // decoder layer 1: input h0 (fresh bf16 in hbf[0])
    GEMMF(hbf, wdecih + (size_t)H3 * H, gi, N, H3, H);
    GEMMF(hbf + (size_t)N * H, wdechh + (size_t)H3 * H, gh, N, H3, H);
    ELEM(k_gru_gates, N * H, gi, gh, dec_bih + H3, dec_bhh + H3,
         hbuf + (size_t)N * H, hbf + (size_t)N * H,
         (__bf16*)nullptr, out + (size_t)t * N * H);        // outs[t]
  }

  // ---- final hidden state hT -> tail of d_out ----
  ELEM(k_copy_f32, (size_t)L * N * H, hbuf, out + (size_t)S * N * H,
       (size_t)L * N * H);

#undef ELEM
#undef GEMMF
#undef GEMMR
}
